// OriginalGeometricRound_8186207666910
// MI455X (gfx1250) — matrix operand order repaired and checked by reading, not compiled, not run on hardware
//
#include <hip/hip_runtime.h>

// ---------------------------------------------------------------------------
// CDNA5 (gfx1250) fused geometric-product + LN + SwiGLU FFN kernel.
// wave32, WMMA bf16 16x16x32, 8 waves / 256 threads per block, 64 rows/block.
// Async global->LDS staging of x tiles (ASYNCcnt), software-pipelined frags.
// ---------------------------------------------------------------------------

typedef __bf16 bf16_t;
typedef __attribute__((ext_vector_type(16))) __bf16 v16bf;
typedef __attribute__((ext_vector_type(8)))  __bf16 v8bf;
typedef __attribute__((ext_vector_type(8)))  float  v8f;
typedef __attribute__((ext_vector_type(4)))  int    v4i_t;

#define D_BLADE 256
#define D_FFN   1024
#define MT      64          // rows (token*8+blade) per block
#define TPB     256
#define AP      264         // a_bf row pitch (bf16 elems; 16B-aligned, bank-skewed)
#define WP      264         // staged W col pitch (K=256 + pad)
#define HP      72          // h row pitch (K=64 + pad)
#define DP      72          // staged W_down col pitch (K=64 + pad)
#define NCHUNK  64          // FFN N-chunk
#define LN_EPS  1e-5f

#if defined(__has_builtin)
#if __has_builtin(__builtin_amdgcn_global_load_async_to_lds_b128)
#define HAS_ASYNC_LDS 1
#endif
#endif
#ifndef HAS_ASYNC_LDS
#define HAS_ASYNC_LDS 0
#endif

// Cl(3) Cayley table, p = i*8 + j  ->  (target blade, sign)
constexpr int TKC[64] = {
    0,1,2,3,4,5,6,7,
    1,0,4,5,2,3,7,6,
    2,4,0,6,1,7,3,5,
    3,5,6,0,7,1,2,4,
    4,2,1,7,0,6,5,3,
    5,3,7,1,6,0,4,2,
    6,7,3,2,5,4,0,1,
    7,6,5,4,3,2,1,0};
constexpr float SGC[64] = {
    1,1,1,1,1,1,1,1,
    1,1,1,1,1,1,1,1,
    1,-1,1,1,-1,-1,1,-1,
    1,-1,-1,1,1,-1,-1,1,
    1,-1,1,1,-1,-1,1,-1,
    1,-1,-1,1,1,-1,-1,1,
    1,1,-1,1,-1,1,-1,-1,
    1,1,-1,1,-1,1,-1,-1};

__device__ __forceinline__ bf16_t f2bf(float f) {
    unsigned u = __builtin_bit_cast(unsigned, f);
    unsigned r = (u + 0x7FFFu + ((u >> 16) & 1u)) >> 16;   // RNE
    unsigned short s = (unsigned short)r;
    return __builtin_bit_cast(bf16_t, s);
}

__device__ __forceinline__ float sigm(float z) { return 1.0f / (1.0f + __expf(-z)); }

union FragU { v16bf v; v8bf h[2]; };

// A fragment (16x32 bf16, row-major source):
// lanes 0-15 : row = mBase+lane,    K = kBase+{0..7, 16..23}
// lanes 16-31: row = mBase+lane-16, K = kBase+{8..15, 24..31}
__device__ __forceinline__ v16bf load_a_frag(const bf16_t* base, int pitch,
                                             int mBase, int kBase, int lane) {
    int row  = mBase + (lane & 15);
    int kOff = (lane < 16) ? 0 : 8;
    const bf16_t* p = base + row * pitch + kBase + kOff;
    FragU f;
    f.h[0] = *(const v8bf*)(p);
    f.h[1] = *(const v8bf*)(p + 16);
    return f.v;
}

// B fragment (32x16 bf16, staged column-major [col][k]):
// lanes 0-15 : col = nBase+lane,    K = kBase+{0..7, 16..23}
// lanes 16-31: col = nBase+lane-16, K = kBase+{8..15, 24..31}
__device__ __forceinline__ v16bf load_b_frag(const bf16_t* base, int pitch,
                                             int nBase, int kBase, int lane) {
    int col  = nBase + (lane & 15);
    int kOff = (lane < 16) ? 0 : 8;
    const bf16_t* p = base + col * pitch + kBase + kOff;
    FragU f;
    f.h[0] = *(const v8bf*)(p);
    f.h[1] = *(const v8bf*)(p + 16);
    return f.v;
}

__device__ __forceinline__ v8f wmma_bf16(v16bf a, v16bf b, v8f c) {
    return __builtin_amdgcn_wmma_f32_16x16x32_bf16(false, a, false, b,
                                                   (short)0, c, false, false);
}

// Copy one token tile (8 blades x 256 f32 = 8 KB) global -> LDS.
// Async path uses GLOBAL_LOAD_ASYNC_TO_LDS_B128 (ASYNCcnt tracked):
// per-lane 16B copy, no VGPR round-trip.
__device__ __forceinline__ void copy_token(const float* __restrict__ g,
                                           float* l, int tid) {
#pragma unroll
    for (int q = 0; q < 2; ++q) {
        const float* gp = g + q * 1024 + tid * 4;   // lanes -> contiguous 16B
        float* lp = l + q * 1024 + tid * 4;
#if HAS_ASYNC_LDS
        __builtin_amdgcn_global_load_async_to_lds_b128(
            (__attribute__((address_space(1))) v4i_t*)gp,
            (__attribute__((address_space(3))) v4i_t*)lp, 0, 0);
#else
        *(float4*)lp = *(const float4*)gp;
#endif
    }
}

__device__ __forceinline__ void wait_async_copies() {
#if HAS_ASYNC_LDS
#if __has_builtin(__builtin_amdgcn_s_wait_asynccnt)
    __builtin_amdgcn_s_wait_asynccnt(0);
#else
    asm volatile("s_wait_asynccnt 0x0" ::: "memory");
#endif
#endif
}

// Phase-disjoint LDS reuse: x double-buffer + token mix buffer (phase 1)
// overlays the W_gate/W_up chunk staging (phase 2).
struct SmemP1 {
    float xtok[2][8 * D_BLADE];   // 16 KB: double-buffered x token tiles
    float tokbuf[8 * D_BLADE];    //  8 KB: gated mix, one token
};
struct SmemP2 {
    bf16_t wg[NCHUNK * WP];       // W_gate chunk, col-major bf16
    bf16_t wu[NCHUNK * WP];       // W_up   chunk, col-major bf16
};
union SmemU { SmemP1 p1; SmemP2 p2; };

__global__ __launch_bounds__(TPB, 1)
void geo_swiglu_kernel(const float* __restrict__ x,
                       const float* __restrict__ iw,
                       const float* __restrict__ gg,
                       const float* __restrict__ lnw,
                       const float* __restrict__ lnb,
                       const float* __restrict__ Wg,
                       const float* __restrict__ Wu,
                       const float* __restrict__ Wd,
                       float* __restrict__ out) {
    __shared__ __align__(16) SmemU  smem;                // 67584 B (union)
    __shared__ __align__(16) bf16_t a_bf[MT * AP];       // 33792 B LN'd acts
    __shared__ __align__(16) bf16_t h_s [MT * HP];       //  9216 B silu(g)*u
    __shared__ __align__(16) bf16_t wd_s[D_BLADE * DP];  // 36864 B W_down chunk
    __shared__ float sws[64];

    const int tid  = threadIdx.x;
    const int lane = tid & 31;
    const int wave = tid >> 5;
    const int row0 = blockIdx.x * MT;                    // first flat row of block

    // ---- Phase 0: stage small params, kick off first async x copy -------
    copy_token(x + (size_t)row0 * D_BLADE, smem.p1.xtok[0], tid);
    if (tid < 64) sws[tid] = SGC[tid] * sigm(iw[tid]);
    const float g = sigm(gg[0]);

    // ---- Phase 1: geometric product + gate + LayerNorm -> a_bf ----------
    for (int t = 0; t < 8; ++t) {
        wait_async_copies();
        __syncthreads();                                 // xtok[t&1] ready block-wide
        if (t + 1 < 8)                                   // overlap next token copy
            copy_token(x + (size_t)(row0 + (t + 1) * 8) * D_BLADE,
                       smem.p1.xtok[(t + 1) & 1], tid);

        const float* xt = smem.p1.xtok[t & 1];
        const int d = tid;                               // 0..255
        float xv[8];
#pragma unroll
        for (int b = 0; b < 8; ++b) xv[b] = xt[b * D_BLADE + d];
        float acc[8] = {0.f, 0.f, 0.f, 0.f, 0.f, 0.f, 0.f, 0.f};
#pragma unroll
        for (int p = 0; p < 64; ++p)
            acc[TKC[p]] += sws[p] * xv[p >> 3] * xv[p & 7];
#pragma unroll
        for (int b = 0; b < 8; ++b)
            smem.p1.tokbuf[b * D_BLADE + d] = g * acc[b] + (1.0f - g) * xv[b];
        __syncthreads();

        // one wave per blade-row: LayerNorm over D=256
        {
            float v[8], s = 0.f, s2 = 0.f;
#pragma unroll
            for (int k = 0; k < 8; ++k) {
                v[k] = smem.p1.tokbuf[wave * D_BLADE + lane + 32 * k];
                s += v[k];
                s2 += v[k] * v[k];
            }
#pragma unroll
            for (int m = 16; m >= 1; m >>= 1) {
                s  += __shfl_xor(s,  m, 32);
                s2 += __shfl_xor(s2, m, 32);
            }
            const float mu = s * (1.0f / D_BLADE);
            const float var = s2 * (1.0f / D_BLADE) - mu * mu;
            const float rs = __frsqrt_rn(var + LN_EPS);
#pragma unroll
            for (int k = 0; k < 8; ++k) {
                const int col = lane + 32 * k;
                const float a = (v[k] - mu) * rs * lnw[col] + lnb[col];
                a_bf[(t * 8 + wave) * AP + col] = f2bf(a);
            }
        }
        __syncthreads();
    }

    // ---- Phase 2: fused SwiGLU FFN (WMMA bf16) ---------------------------
    const int mSub  = wave & 3;      // 4 row stripes of 16
    const int nHalf = wave >> 2;     // 2 halves
    v8f oacc[8];
#pragma unroll
    for (int tn = 0; tn < 8; ++tn) oacc[tn] = (v8f){0.f,0.f,0.f,0.f,0.f,0.f,0.f,0.f};

    for (int nc = 0; nc < D_FFN / NCHUNK; ++nc) {
        const int n0 = nc * NCHUNK;
        // stage W_gate / W_up chunk (col-major bf16) into the unioned region
        {
            const int c = tid & 63, kr = tid >> 6;
#pragma unroll 4
            for (int k = kr; k < D_BLADE; k += 4) {
                smem.p2.wg[c * WP + k] = f2bf(Wg[(size_t)k * D_FFN + n0 + c]);
                smem.p2.wu[c * WP + k] = f2bf(Wu[(size_t)k * D_FFN + n0 + c]);
            }
            if (nc + 1 < D_FFN / NCHUNK) {   // speculative stream of next chunk
                __builtin_prefetch(&Wg[(size_t)kr * D_FFN + n0 + NCHUNK + c], 0, 1);
                __builtin_prefetch(&Wu[(size_t)kr * D_FFN + n0 + NCHUNK + c], 0, 1);
            }
        }
        // stage W_down chunk (col-major bf16): wd_s[col][k], k = 0..63
        {
#pragma unroll 4
            for (int k = 0; k < NCHUNK; ++k)
                wd_s[tid * DP + k] = f2bf(Wd[(size_t)(n0 + k) * D_BLADE + tid]);
        }
        __syncthreads();

        // GEMM 1: h = silu(A @ Wg) * (A @ Wu), per-wave 16x32 subtile.
        // A fragment pipelined one k-step ahead; B loads grouped before WMMAs.
        v8f hg[2], hu[2];
#pragma unroll
        for (int tn = 0; tn < 2; ++tn) {
            hg[tn] = (v8f){0.f,0.f,0.f,0.f,0.f,0.f,0.f,0.f};
            hu[tn] = (v8f){0.f,0.f,0.f,0.f,0.f,0.f,0.f,0.f};
        }
        const int nb0 = nHalf * 32, nb1 = nb0 + 16;
        v16bf af = load_a_frag(a_bf, AP, mSub * 16, 0, lane);
#pragma unroll
        for (int kk = 0; kk < D_BLADE; kk += 32) {
            const v16bf b0 = load_b_frag(smem.p2.wg, WP, nb0, kk, lane);
            const v16bf b1 = load_b_frag(smem.p2.wu, WP, nb0, kk, lane);
            const v16bf b2 = load_b_frag(smem.p2.wg, WP, nb1, kk, lane);
            const v16bf b3 = load_b_frag(smem.p2.wu, WP, nb1, kk, lane);
            v16bf afn = af;
            if (kk + 32 < D_BLADE)
                afn = load_a_frag(a_bf, AP, mSub * 16, kk + 32, lane);
            hg[0] = wmma_bf16(af, b0, hg[0]);
            hu[0] = wmma_bf16(af, b1, hu[0]);
            hg[1] = wmma_bf16(af, b2, hg[1]);
            hu[1] = wmma_bf16(af, b3, hu[1]);
            af = afn;
        }
        // SiLU * up, write bf16 h tile (C/D layout: VGPR r, M = r + 8*(lane/16), N = lane%16)
#pragma unroll
        for (int tn = 0; tn < 2; ++tn) {
#pragma unroll
            for (int r = 0; r < 8; ++r) {
                const float gz = hg[tn][r];
                const float hv = gz * sigm(gz) * hu[tn][r];
                const int M = r + 8 * (lane >> 4);
                const int N = lane & 15;
                h_s[(mSub * 16 + M) * HP + nHalf * 32 + tn * 16 + N] = f2bf(hv);
            }
        }
        __syncthreads();

        // GEMM 2: out += h_chunk @ Wd_chunk, per-wave 16x128 accumulators.
        // All 8 B fragments issued before the WMMA burst of each k-step.
#pragma unroll
        for (int kk = 0; kk < NCHUNK; kk += 32) {
            const v16bf ah = load_a_frag(h_s, HP, mSub * 16, kk, lane);
            v16bf bfr[8];
#pragma unroll
            for (int tn = 0; tn < 8; ++tn)
                bfr[tn] = load_b_frag(wd_s, DP, nHalf * 128 + tn * 16, kk, lane);
#pragma unroll
            for (int tn = 0; tn < 8; ++tn)
                oacc[tn] = wmma_bf16(ah, bfr[tn], oacc[tn]);
        }
        __syncthreads();
    }

    // ---- Final: residual add + store ------------------------------------
#pragma unroll
    for (int tn = 0; tn < 8; ++tn) {
        const int nb = nHalf * 128 + tn * 16;
#pragma unroll
        for (int r = 0; r < 8; ++r) {
            const int M = r + 8 * (lane >> 4);
            const size_t grow = (size_t)row0 + mSub * 16 + M;
            const int col = nb + (lane & 15);
            out[grow * D_BLADE + col] = oacc[tn][r] + x[grow * D_BLADE + col];
        }
    }
}

extern "C" void kernel_launch(void* const* d_in, const int* in_sizes, int n_in,
                              void* d_out, int out_size, void* d_ws, size_t ws_size,
                              hipStream_t stream) {
    const float* x   = (const float*)d_in[0];
    const float* iw  = (const float*)d_in[1];
    const float* gg  = (const float*)d_in[2];
    const float* lnw = (const float*)d_in[3];
    const float* lnb = (const float*)d_in[4];
    const float* Wg  = (const float*)d_in[5];
    const float* Wu  = (const float*)d_in[6];
    const float* Wd  = (const float*)d_in[7];
    float* out = (float*)d_out;

    const int rows = in_sizes[0] / D_BLADE;   // 65536 flat rows
    const int grid = rows / MT;               // 1024 blocks
    hipLaunchKernelGGL(geo_swiglu_kernel, dim3(grid), dim3(TPB), 0, stream,
                       x, iw, gg, lnw, lnb, Wg, Wu, Wd, out);
}
